// QuantumBiologicalGNN_19670950215945
// MI455X (gfx1250) — matrix operand order repaired and checked by reading, not compiled, not run on hardware
//
#include <hip/hip_runtime.h>
#include <hip/hip_bf16.h>

// ---- problem constants (match reference) ----
#define Nn 4096
#define En 65536
#define INd 512
#define Hd 256
#define QDd 256
#define NHh 8
#define HDd 32
#define OUTd 10
#define Ld 2
#define CTt 50

typedef __attribute__((ext_vector_type(8))) float v8f;
typedef __attribute__((ext_vector_type(16))) __bf16 v16bf;

__device__ __forceinline__ __bf16 tobf(float f) { return (__bf16)f; }
// A-matrix (16x32 bf16) lane m=l&15, h=l>>4: element e -> K offset
__device__ __forceinline__ int aKi(int e, int h) { return e + 16 * h; }
// B-matrix (32x16 bf16) lane n=l&15, h=l>>4: element e -> K offset
__device__ __forceinline__ int bKi(int e, int h) { return e + 16 * h; }

__device__ __forceinline__ v8f wmma_bf16(v16bf a, v16bf b, v8f c) {
  return __builtin_amdgcn_wmma_f32_16x16x32_bf16(false, a, false, b, (short)0, c,
                                                 false, false);
}

// load A fragment: 16 bf16 from one contiguous 16-float run of a row
__device__ __forceinline__ v16bf load_afrag(const float* __restrict__ rowk, int h) {
  // rowk points at row + k0 ; lane uses [16h..16h+15]
  const float4 a0 = *(const float4*)(rowk + 16 * h);
  const float4 a1 = *(const float4*)(rowk + 16 * h + 4);
  const float4 a2 = *(const float4*)(rowk + 16 * h + 8);
  const float4 a3 = *(const float4*)(rowk + 16 * h + 12);
  v16bf a;
  a[0] = tobf(a0.x); a[1] = tobf(a0.y); a[2] = tobf(a0.z); a[3] = tobf(a0.w);
  a[4] = tobf(a1.x); a[5] = tobf(a1.y); a[6] = tobf(a1.z); a[7] = tobf(a1.w);
  a[8] = tobf(a2.x); a[9] = tobf(a2.y); a[10] = tobf(a2.z); a[11] = tobf(a2.w);
  a[12] = tobf(a3.x); a[13] = tobf(a3.y); a[14] = tobf(a3.z); a[15] = tobf(a3.w);
  return a;
}

// stage a 32x256 f32 row-block (ld=256) into LDS as bf16, coalesced float4
__device__ __forceinline__ void stage_256(const float* __restrict__ W,
                                          __bf16* __restrict__ lds, int lane) {
  for (int idx = lane; idx < 2048; idx += 32) {
    const int r = idx >> 6, c4 = idx & 63;
    const float4 v = ((const float4*)(W + r * 256))[c4];
    __bf16* d = lds + r * 256 + c4 * 4;
    d[0] = tobf(v.x); d[1] = tobf(v.y); d[2] = tobf(v.z); d[3] = tobf(v.w);
  }
}

// =====================================================================
// Generic GEMM: C[M,N] = act(A[M,K](f32) @ B + bias), bf16 WMMA, f32 accum
// TRANSB==0: B[k*ldb+n] ; TRANSB==1: B[n*ldb+k]
// 1 wave = one 16x16 tile. grid = (N/16, M/16), block = 32
// =====================================================================
template <int TRANSB>
__global__ __launch_bounds__(32) void gemm_wmma(
    const float* __restrict__ A, int lda, const float* __restrict__ B, int ldb,
    const float* __restrict__ bias, float* __restrict__ C, int ldc, int K,
    int relu) {
  __shared__ __bf16 ldsB[32 * 16];
  const int n0 = blockIdx.x * 16;
  const int m0 = blockIdx.y * 16;
  const int lane = threadIdx.x;
  const int m = lane & 15, h = lane >> 4;
  const float* Arow = A + (m0 + m) * lda;
  v8f acc = {};
  for (int k0 = 0; k0 < K; k0 += 32) {
    if (TRANSB) {
      // B tile: rows n (contiguous in k). 512 elems = 128 float4
      for (int idx = lane; idx < 128; idx += 32) {
        const int n = idx >> 3, k4 = (idx & 7) * 4;
        const float4 v = *(const float4*)(B + (n0 + n) * ldb + k0 + k4);
        ldsB[(k4 + 0) * 16 + n] = tobf(v.x);
        ldsB[(k4 + 1) * 16 + n] = tobf(v.y);
        ldsB[(k4 + 2) * 16 + n] = tobf(v.z);
        ldsB[(k4 + 3) * 16 + n] = tobf(v.w);
      }
    } else {
      // B tile: rows k (contiguous in n)
      for (int idx = lane; idx < 128; idx += 32) {
        const int k = idx >> 2, c4 = (idx & 3) * 4;
        const float4 v = *(const float4*)(B + (k0 + k) * ldb + n0 + c4);
        __bf16* d = ldsB + k * 16 + c4;
        d[0] = tobf(v.x); d[1] = tobf(v.y); d[2] = tobf(v.z); d[3] = tobf(v.w);
      }
    }
    const v16bf a = load_afrag(Arow + k0, h);
    v16bf b;
#pragma unroll
    for (int e = 0; e < 16; ++e) b[e] = ldsB[bKi(e, h) * 16 + m];
    acc = wmma_bf16(a, b, acc);
  }
  const int col = n0 + m;
  const float bb = bias ? bias[col] : 0.f;
#pragma unroll
  for (int r = 0; r < 8; ++r) {
    const int row = m0 + r + 8 * h;
    float v = acc[r] + bb;
    if (relu) v = fmaxf(v, 0.f);
    C[row * ldc + col] = v;
  }
}

// =====================================================================
// Build dense adjacency bitmask: bit(src*N + tgt)
// =====================================================================
__global__ __launch_bounds__(256) void build_mask(const int* __restrict__ src,
                                                  const int* __restrict__ tgt,
                                                  unsigned* __restrict__ bits) {
  const int i = blockIdx.x * 256 + threadIdx.x;
  if (i < En) {
    const unsigned p = (unsigned)src[i] * (unsigned)Nn + (unsigned)tgt[i];
    atomicOr(&bits[p >> 5], 1u << (p & 31));
  }
}

// =====================================================================
// Q += 0.1*cemb[ct], K += 0.1*cemb[ct]
// =====================================================================
__global__ __launch_bounds__(256) void add_cemb(float* __restrict__ Q,
                                                float* __restrict__ Kx,
                                                const float* __restrict__ cemb,
                                                const int* __restrict__ ct) {
  const int row = blockIdx.x, c = threadIdx.x;
  const float cv = 0.1f * cemb[ct[row] * QDd + c];
  Q[row * QDd + c] += cv;
  Kx[row * QDd + c] += cv;
}

// =====================================================================
// Dense graph-masked flash attention: O = softmax(mask(QK^T/16)) @ V
// 1 wave = 16 rows x full 256 cols. grid = N/16, block = 32
// =====================================================================
__global__ __launch_bounds__(32) void graph_attn(
    const float* __restrict__ Q, const float* __restrict__ Km,
    const float* __restrict__ Vm, const unsigned* __restrict__ bits,
    float* __restrict__ O) {
  __shared__ __bf16 ldsK[32 * 256];
  __shared__ __bf16 ldsV[32 * 256];
  __shared__ __bf16 ldsP[16 * 32];
  const int m0 = blockIdx.x * 16;
  const int lane = threadIdx.x;
  const int mm = lane & 15, hh = lane >> 4;

  v16bf qf[8];
  {
    const float* qrow = Q + (m0 + mm) * QDd;
#pragma unroll
    for (int kk = 0; kk < 8; ++kk) qf[kk] = load_afrag(qrow + kk * 32, hh);
  }

  v8f oacc[16];
#pragma unroll
  for (int nt = 0; nt < 16; ++nt) oacc[nt] = (v8f){};
  float mrun[8], lrun[8];
#pragma unroll
  for (int r = 0; r < 8; ++r) { mrun[r] = -3.0e38f; lrun[r] = 0.f; }

  for (int j0 = 0; j0 < Nn; j0 += 32) {
    stage_256(Km + j0 * QDd, ldsK, lane);
    stage_256(Vm + j0 * QDd, ldsV, lane);
    v8f s[2];
#pragma unroll
    for (int t = 0; t < 2; ++t) {
      s[t] = (v8f){};
#pragma unroll
      for (int kk = 0; kk < 8; ++kk) {
        v16bf bfr;
#pragma unroll
        for (int e = 0; e < 16; ++e)
          bfr[e] = ldsK[(t * 16 + mm) * 256 + kk * 32 + bKi(e, hh)];
        s[t] = wmma_bf16(qf[kk], bfr, s[t]);
      }
    }
    const float scale = 0.0625f;  // 1/sqrt(256)
    float corr[8];
#pragma unroll
    for (int r = 0; r < 8; ++r) {
      const int row = m0 + r + 8 * hh;
      // both lane bits (j0+mm, j0+16+mm) live in one aligned 32-bit word
      const unsigned w = bits[row * (Nn / 32) + (j0 >> 5)];
      float s0 = s[0][r] * scale, s1 = s[1][r] * scale;
      if (!((w >> mm) & 1u)) s0 = -1e9f;
      if (!((w >> (mm + 16)) & 1u)) s1 = -1e9f;
      float rmax = fmaxf(s0, s1);
#pragma unroll
      for (int off = 1; off < 16; off <<= 1)
        rmax = fmaxf(rmax, __shfl_xor(rmax, off, 32));
      const float nm = fmaxf(mrun[r], rmax);
      const float e0 = __expf(s0 - nm), e1 = __expf(s1 - nm);
      float rsum = e0 + e1;
#pragma unroll
      for (int off = 1; off < 16; off <<= 1) rsum += __shfl_xor(rsum, off, 32);
      const float c = __expf(mrun[r] - nm);
      lrun[r] = lrun[r] * c + rsum;
      mrun[r] = nm;
      corr[r] = c;
      ldsP[(r + 8 * hh) * 32 + mm] = tobf(e0);
      ldsP[(r + 8 * hh) * 32 + 16 + mm] = tobf(e1);
    }
#pragma unroll
    for (int nt = 0; nt < 16; ++nt)
#pragma unroll
      for (int r = 0; r < 8; ++r) oacc[nt][r] *= corr[r];
    v16bf pf;
#pragma unroll
    for (int e = 0; e < 16; ++e) pf[e] = ldsP[mm * 32 + aKi(e, hh)];
#pragma unroll
    for (int nt = 0; nt < 16; ++nt) {
      v16bf bfr;
#pragma unroll
      for (int e = 0; e < 16; ++e)
        bfr[e] = ldsV[bKi(e, hh) * 256 + nt * 16 + mm];
      oacc[nt] = wmma_bf16(pf, bfr, oacc[nt]);
    }
  }
#pragma unroll
  for (int nt = 0; nt < 16; ++nt)
#pragma unroll
    for (int r = 0; r < 8; ++r) {
      const int row = m0 + r + 8 * hh;
      O[row * QDd + nt * 16 + mm] = oacc[nt][r] / lrun[r];
    }
}

// =====================================================================
// Per-head flash MHA (no mask). grid = (N/16, NH), block = 32
// =====================================================================
__global__ __launch_bounds__(32) void mha_attn(const float* __restrict__ q,
                                               const float* __restrict__ k,
                                               const float* __restrict__ v,
                                               float* __restrict__ o) {
  __shared__ __bf16 ldsP[16 * 32];
  const int m0 = blockIdx.x * 16;
  const int hoff = blockIdx.y * HDd;
  const int lane = threadIdx.x;
  const int mm = lane & 15, hh = lane >> 4;

  const v16bf qf = load_afrag(q + (m0 + mm) * QDd + hoff, hh);

  v8f oacc[2];
  oacc[0] = (v8f){}; oacc[1] = (v8f){};
  float mrun[8], lrun[8];
#pragma unroll
  for (int r = 0; r < 8; ++r) { mrun[r] = -3.0e38f; lrun[r] = 0.f; }

  for (int j0 = 0; j0 < Nn; j0 += 32) {
    v8f s[2];
#pragma unroll
    for (int t = 0; t < 2; ++t) {
      // B fragment = 16 contiguous floats of a K row
      const float* kr = k + (j0 + t * 16 + mm) * QDd + hoff + 16 * hh;
      const float4 b0 = *(const float4*)(kr);
      const float4 b1 = *(const float4*)(kr + 4);
      const float4 b2 = *(const float4*)(kr + 8);
      const float4 b3 = *(const float4*)(kr + 12);
      v16bf bfr;
      bfr[0] = tobf(b0.x); bfr[1] = tobf(b0.y); bfr[2] = tobf(b0.z); bfr[3] = tobf(b0.w);
      bfr[4] = tobf(b1.x); bfr[5] = tobf(b1.y); bfr[6] = tobf(b1.z); bfr[7] = tobf(b1.w);
      bfr[8] = tobf(b2.x); bfr[9] = tobf(b2.y); bfr[10] = tobf(b2.z); bfr[11] = tobf(b2.w);
      bfr[12] = tobf(b3.x); bfr[13] = tobf(b3.y); bfr[14] = tobf(b3.z); bfr[15] = tobf(b3.w);
      s[t] = wmma_bf16(qf, bfr, (v8f){});
    }
    const float scale = 0.17677669529663688f;  // 1/sqrt(32)
    float corr[8];
#pragma unroll
    for (int r = 0; r < 8; ++r) {
      float s0 = s[0][r] * scale, s1 = s[1][r] * scale;
      float rmax = fmaxf(s0, s1);
#pragma unroll
      for (int off = 1; off < 16; off <<= 1)
        rmax = fmaxf(rmax, __shfl_xor(rmax, off, 32));
      const float nm = fmaxf(mrun[r], rmax);
      const float e0 = __expf(s0 - nm), e1 = __expf(s1 - nm);
      float rsum = e0 + e1;
#pragma unroll
      for (int off = 1; off < 16; off <<= 1) rsum += __shfl_xor(rsum, off, 32);
      const float c = __expf(mrun[r] - nm);
      lrun[r] = lrun[r] * c + rsum;
      mrun[r] = nm;
      corr[r] = c;
      ldsP[(r + 8 * hh) * 32 + mm] = tobf(e0);
      ldsP[(r + 8 * hh) * 32 + 16 + mm] = tobf(e1);
    }
#pragma unroll
    for (int nt = 0; nt < 2; ++nt)
#pragma unroll
      for (int r = 0; r < 8; ++r) oacc[nt][r] *= corr[r];
    v16bf pf;
#pragma unroll
    for (int e = 0; e < 16; ++e) pf[e] = ldsP[mm * 32 + aKi(e, hh)];
#pragma unroll
    for (int nt = 0; nt < 2; ++nt) {
      v16bf bfr;
#pragma unroll
      for (int e = 0; e < 16; ++e)
        bfr[e] = tobf(v[(j0 + bKi(e, hh)) * QDd + hoff + nt * 16 + mm]);
      oacc[nt] = wmma_bf16(pf, bfr, oacc[nt]);
    }
  }
#pragma unroll
  for (int nt = 0; nt < 2; ++nt)
#pragma unroll
    for (int r = 0; r < 8; ++r) {
      const int row = m0 + r + 8 * hh;
      o[row * QDd + hoff + nt * 16 + mm] = oacc[nt][r] / lrun[r];
    }
}

// =====================================================================
// Fused edge message MLP + scatter-add. grid = E/16, block = 32
// m = relu(relu([h_tgt||h_src]@Wm1+b1)@Wm2+b2)@Wm3+b3 ; AGG[tgt] += m
// =====================================================================
__global__ __launch_bounds__(32) void msg_mlp(
    const float* __restrict__ h, const int* __restrict__ src,
    const int* __restrict__ tgt, const float* __restrict__ Wm1,
    const float* __restrict__ bm1, const float* __restrict__ Wm2,
    const float* __restrict__ bm2, const float* __restrict__ Wm3,
    const float* __restrict__ bm3, float* __restrict__ AGG) {
  __shared__ __bf16 ldsA[16 * 512];
  __shared__ __bf16 ldsW[32 * 256];
  __shared__ __bf16 ldsM1[16 * 256];
  __shared__ __bf16 ldsM2[16 * 256];
  const int e0 = blockIdx.x * 16;
  const int lane = threadIdx.x;
  const int mm = lane & 15, hh = lane >> 4;

  // stage gathered A = [h[tgt] || h[src]] as bf16, vectorized
  for (int idx = lane; idx < 2048; idx += 32) {
    const int r = idx >> 7, c4 = idx & 127;
    const int node = (c4 < 64) ? tgt[e0 + r] : src[e0 + r];
    const float4 v = ((const float4*)(h + node * Hd))[c4 & 63];
    __bf16* d = ldsA + r * 512 + c4 * 4;
    d[0] = tobf(v.x); d[1] = tobf(v.y); d[2] = tobf(v.z); d[3] = tobf(v.w);
  }

  v8f acc[16];
  // ---- stage 1: K=512, relu -> ldsM1 ----
#pragma unroll
  for (int nt = 0; nt < 16; ++nt) acc[nt] = (v8f){};
  for (int kk = 0; kk < 16; ++kk) {
    stage_256(Wm1 + kk * 32 * QDd, ldsW, lane);
    v16bf a;
#pragma unroll
    for (int e = 0; e < 16; ++e) a[e] = ldsA[mm * 512 + kk * 32 + aKi(e, hh)];
#pragma unroll
    for (int nt = 0; nt < 16; ++nt) {
      v16bf b;
#pragma unroll
      for (int e = 0; e < 16; ++e) b[e] = ldsW[bKi(e, hh) * 256 + nt * 16 + mm];
      acc[nt] = wmma_bf16(a, b, acc[nt]);
    }
  }
#pragma unroll
  for (int nt = 0; nt < 16; ++nt) {
    const float bb = bm1[nt * 16 + mm];
#pragma unroll
    for (int r = 0; r < 8; ++r)
      ldsM1[(r + 8 * hh) * 256 + nt * 16 + mm] = tobf(fmaxf(acc[nt][r] + bb, 0.f));
  }
  // ---- stage 2: K=256, relu -> ldsM2 ----
#pragma unroll
  for (int nt = 0; nt < 16; ++nt) acc[nt] = (v8f){};
  for (int kk = 0; kk < 8; ++kk) {
    stage_256(Wm2 + kk * 32 * QDd, ldsW, lane);
    v16bf a;
#pragma unroll
    for (int e = 0; e < 16; ++e) a[e] = ldsM1[mm * 256 + kk * 32 + aKi(e, hh)];
#pragma unroll
    for (int nt = 0; nt < 16; ++nt) {
      v16bf b;
#pragma unroll
      for (int e = 0; e < 16; ++e) b[e] = ldsW[bKi(e, hh) * 256 + nt * 16 + mm];
      acc[nt] = wmma_bf16(a, b, acc[nt]);
    }
  }
#pragma unroll
  for (int nt = 0; nt < 16; ++nt) {
    const float bb = bm2[nt * 16 + mm];
#pragma unroll
    for (int r = 0; r < 8; ++r)
      ldsM2[(r + 8 * hh) * 256 + nt * 16 + mm] = tobf(fmaxf(acc[nt][r] + bb, 0.f));
  }
  // ---- stage 3: K=256 -> scatter add ----
#pragma unroll
  for (int nt = 0; nt < 16; ++nt) acc[nt] = (v8f){};
  for (int kk = 0; kk < 8; ++kk) {
    stage_256(Wm3 + kk * 32 * Hd, ldsW, lane);
    v16bf a;
#pragma unroll
    for (int e = 0; e < 16; ++e) a[e] = ldsM2[mm * 256 + kk * 32 + aKi(e, hh)];
#pragma unroll
    for (int nt = 0; nt < 16; ++nt) {
      v16bf b;
#pragma unroll
      for (int e = 0; e < 16; ++e) b[e] = ldsW[bKi(e, hh) * 256 + nt * 16 + mm];
      acc[nt] = wmma_bf16(a, b, acc[nt]);
    }
  }
#pragma unroll
  for (int nt = 0; nt < 16; ++nt) {
    const float bb = bm3[nt * 16 + mm];
#pragma unroll
    for (int r = 0; r < 8; ++r) {
      const int t = tgt[e0 + r + 8 * hh];
      atomicAdd(&AGG[t * Hd + nt * 16 + mm], acc[nt][r] + bb);
    }
  }
}

// =====================================================================
// out = LayerNorm(res + agg + mha + ga). grid = N, block = 256
// =====================================================================
__global__ __launch_bounds__(256) void add_ln(
    const float* __restrict__ res, const float* __restrict__ agg,
    const float* __restrict__ mha, const float* __restrict__ ga,
    const float* __restrict__ g, const float* __restrict__ b,
    float* __restrict__ out) {
  __shared__ float s1[256], s2[256];
  const int row = blockIdx.x, c = threadIdx.x;
  const int i = row * Hd + c;
  const float t = res[i] + agg[i] + mha[i] + ga[i];
  s1[c] = t; s2[c] = t * t;
  __syncthreads();
  for (int st = 128; st > 0; st >>= 1) {
    if (c < st) { s1[c] += s1[c + st]; s2[c] += s2[c + st]; }
    __syncthreads();
  }
  const float mu = s1[0] * (1.f / Hd);
  const float var = s2[0] * (1.f / Hd) - mu * mu;
  out[i] = (t - mu) * rsqrtf(var + 1e-5f) * g[c] + b[c];
}

// =====================================================================
// pooled = [mean(h,axis0) || max(h,axis0)]. grid = 256 cols, block = 256
// =====================================================================
__global__ __launch_bounds__(256) void pool_mm(const float* __restrict__ hm,
                                               float* __restrict__ pooled) {
  __shared__ float ss[256], sm[256];
  const int c = blockIdx.x, t = threadIdx.x;
  float s = 0.f, mx = -3.0e38f;
  for (int r = t; r < Nn; r += 256) {
    const float v = hm[r * Hd + c];
    s += v; mx = fmaxf(mx, v);
  }
  ss[t] = s; sm[t] = mx;
  __syncthreads();
  for (int st = 128; st > 0; st >>= 1) {
    if (t < st) { ss[t] += ss[t + st]; sm[t] = fmaxf(sm[t], sm[t + st]); }
    __syncthreads();
  }
  if (t == 0) { pooled[c] = ss[0] * (1.f / Nn); pooled[Hd + c] = sm[0]; }
}

// =====================================================================
// out = relu(pooled@Wc1+bc1)@Wc2+bc2. grid = 1, block = 256
// =====================================================================
__global__ __launch_bounds__(256) void classify(
    const float* __restrict__ pooled, const float* __restrict__ Wc1,
    const float* __restrict__ bc1, const float* __restrict__ Wc2,
    const float* __restrict__ bc2, float* __restrict__ out) {
  __shared__ float z[256];
  const int c = threadIdx.x;
  float a = bc1[c];
  for (int k = 0; k < 2 * Hd; ++k) a += pooled[k] * Wc1[k * Hd + c];
  z[c] = fmaxf(a, 0.f);
  __syncthreads();
  if (c < OUTd) {
    float o = bc2[c];
    for (int k = 0; k < Hd; ++k) o += z[k] * Wc2[k * OUTd + c];
    out[c] = o;
  }
}

// =====================================================================
extern "C" void kernel_launch(void* const* d_in, const int* in_sizes, int n_in,
                              void* d_out, int out_size, void* d_ws,
                              size_t ws_size, hipStream_t stream) {
  (void)in_sizes; (void)n_in; (void)out_size; (void)ws_size;
  const float* x     = (const float*)d_in[0];
  const int*   ei    = (const int*)d_in[1];
  const int*   ct    = (const int*)d_in[2];
  const float* Wi    = (const float*)d_in[3];
  const float* bi    = (const float*)d_in[4];
  const float* Wq    = (const float*)d_in[5];
  const float* bq    = (const float*)d_in[6];
  const float* Wk    = (const float*)d_in[7];
  const float* bk    = (const float*)d_in[8];
  const float* Wv    = (const float*)d_in[9];
  const float* bv    = (const float*)d_in[10];
  const float* Bc    = (const float*)d_in[11];
  const float* cemb  = (const float*)d_in[12];
  const float* Win   = (const float*)d_in[13];
  const float* binp  = (const float*)d_in[14];
  const float* Wo    = (const float*)d_in[15];
  const float* bo    = (const float*)d_in[16];
  const float* Wm1   = (const float*)d_in[17];
  const float* bm1   = (const float*)d_in[18];
  const float* Wm2   = (const float*)d_in[19];
  const float* bm2   = (const float*)d_in[20];
  const float* Wm3   = (const float*)d_in[21];
  const float* bm3   = (const float*)d_in[22];
  const float* lng   = (const float*)d_in[23];
  const float* lnb   = (const float*)d_in[24];
  const float* Wc1   = (const float*)d_in[25];
  const float* bc1   = (const float*)d_in[26];
  const float* Wc2   = (const float*)d_in[27];
  const float* bc2   = (const float*)d_in[28];

  const int* src = ei;
  const int* tgt = ei + En;

  const size_t NB = (size_t)Nn * Hd;  // 1M floats per activation buffer
  float* ws   = (float*)d_ws;
  float* hA   = ws + 0 * NB;
  float* hB   = ws + 1 * NB;
  float* Q0   = ws + 2 * NB;  // reused as q
  float* K0   = ws + 3 * NB;  // reused as k
  float* Vb   = ws + 4 * NB;
  float* Q1   = ws + 5 * NB;
  float* K1   = ws + 6 * NB;
  float* GA   = ws + 7 * NB;
  float* v2   = ws + 8 * NB;
  float* AO   = ws + 9 * NB;
  float* MHA  = ws + 10 * NB;
  float* AGG  = ws + 11 * NB;
  unsigned* maskbits = (unsigned*)(ws + 12 * NB);
  const size_t MASKW = (size_t)Nn * Nn / 32;  // 524288 words
  float* pooled = ws + 12 * NB + MASKW;

  const dim3 blk32(32), blk256(256);
  const dim3 gN16(QDd / 16, Nn / 16);

  // adjacency bitmask
  hipMemsetAsync(maskbits, 0, MASKW * sizeof(unsigned), stream);
  build_mask<<<En / 256, blk256, 0, stream>>>(src, tgt, maskbits);

  // h = relu(x @ Wi + bi)
  gemm_wmma<0><<<gN16, blk32, 0, stream>>>(x, INd, Wi, Hd, bi, hA, Hd, INd, 1);

  float* hcur = hA;
  float* hnxt = hB;
  for (int l = 0; l < Ld; ++l) {
    const float* Wq_l = Wq + (size_t)l * Hd * QDd;
    const float* Wk_l = Wk + (size_t)l * Hd * QDd;
    const float* Wv_l = Wv + (size_t)l * Hd * QDd;
    const float* Bc_l = Bc + (size_t)l * QDd * QDd;
    const float* ce_l = cemb + (size_t)l * CTt * QDd;
    const float* Wi_l = Win + (size_t)l * QDd * 3 * QDd;
    const float* bi_l = binp + (size_t)l * 3 * QDd;
    const float* Wo_l = Wo + (size_t)l * QDd * QDd;
    const float* Wm1_l = Wm1 + (size_t)l * (2 * Hd + QDd) * QDd;
    const float* Wm2_l = Wm2 + (size_t)l * QDd * QDd;
    const float* Wm3_l = Wm3 + (size_t)l * QDd * Hd;

    // Q0/K0/V projections
    gemm_wmma<0><<<gN16, blk32, 0, stream>>>(hcur, Hd, Wq_l, QDd, bq + l * QDd, Q0, QDd, Hd, 0);
    gemm_wmma<0><<<gN16, blk32, 0, stream>>>(hcur, Hd, Wk_l, QDd, bk + l * QDd, K0, QDd, Hd, 0);
    gemm_wmma<0><<<gN16, blk32, 0, stream>>>(hcur, Hd, Wv_l, QDd, bv + l * QDd, Vb, QDd, Hd, 0);
    // biological constraint: Q1 = Q0@Bc, K1 = K0@Bc^T
    gemm_wmma<0><<<gN16, blk32, 0, stream>>>(Q0, QDd, Bc_l, QDd, nullptr, Q1, QDd, QDd, 0);
    gemm_wmma<1><<<gN16, blk32, 0, stream>>>(K0, QDd, Bc_l, QDd, nullptr, K1, QDd, QDd, 0);
    // cell-type embedding
    add_cemb<<<Nn, blk256, 0, stream>>>(Q1, K1, ce_l, ct);
    // dense graph-masked attention
    graph_attn<<<Nn / 16, blk32, 0, stream>>>(Q1, K1, Vb, maskbits, GA);
    // MHA input projections (views into Win, ldb = 3*QD)
    gemm_wmma<0><<<gN16, blk32, 0, stream>>>(Q1, QDd, Wi_l + 0 * QDd, 3 * QDd, bi_l + 0 * QDd, Q0, QDd, QDd, 0);
    gemm_wmma<0><<<gN16, blk32, 0, stream>>>(K1, QDd, Wi_l + 1 * QDd, 3 * QDd, bi_l + 1 * QDd, K0, QDd, QDd, 0);
    gemm_wmma<0><<<gN16, blk32, 0, stream>>>(Vb, QDd, Wi_l + 2 * QDd, 3 * QDd, bi_l + 2 * QDd, v2, QDd, QDd, 0);
    // per-head attention + output projection
    mha_attn<<<dim3(Nn / 16, NHh), blk32, 0, stream>>>(Q0, K0, v2, AO);
    gemm_wmma<0><<<gN16, blk32, 0, stream>>>(AO, QDd, Wo_l, QDd, bo + l * QDd, MHA, QDd, QDd, 0);
    // edge message MLP + scatter-add
    hipMemsetAsync(AGG, 0, NB * sizeof(float), stream);
    msg_mlp<<<En / 16, blk32, 0, stream>>>(hcur, src, tgt, Wm1_l, bm1 + l * QDd,
                                           Wm2_l, bm2 + l * QDd, Wm3_l,
                                           bm3 + l * Hd, AGG);
    // residual add + layernorm
    add_ln<<<Nn, blk256, 0, stream>>>(hcur, AGG, MHA, GA, lng + l * Hd,
                                      lnb + l * Hd, hnxt);
    float* tmp = hcur; hcur = hnxt; hnxt = tmp;
  }

  // readout
  pool_mm<<<Hd, blk256, 0, stream>>>(hcur, pooled);
  classify<<<1, blk256, 0, stream>>>(pooled, Wc1, bc1, Wc2, bc2, (float*)d_out);
}
